// ContrastLoss_90907277787777
// MI455X (gfx1250) — matrix operand order repaired and checked by reading, not compiled, not run on hardware
//
#include <hip/hip_runtime.h>
#include <math.h>
#include <stdint.h>

typedef __attribute__((ext_vector_type(16))) _Float16 v16h;
typedef __attribute__((ext_vector_type(8)))  float    v8f;
typedef _Float16 half_t;
typedef unsigned long long u64;
typedef unsigned int u32;

#define BATCH   2
#define NCLS    19
#define HW      25600       // 160*160
#define CH      256
#define NM      3
#define NKEY    2           // NM-1
#define NPAD    1024        // worst-case n (MAX_SAMPLE)
#define KROWS   2048        // NPAD * NKEY
#define LDC2    3072        // NPAD + KROWS
#define MAXS    1024
#define THRESH  100
#define INV_TEMP 10.0f
#define T_OVER_BT (0.1f/0.07f)
#define FEAT_W_  0.1f
#define MASK_W_  0.1f
#define EPS_     1e-16f

// ---- workspace layout (byte offsets) ----
#define OFF_META   0u          // 4096 B of ints: [0]=n_total [1]=total [2]=n_view
                               // [8..46)=cnt [64..102)=hardcnt
                               // [128 + bin*4 + {valid,nhk,nek,base}]
                               // float acc[2] at int index 320
#define OFF_SELB   4096u
#define OFF_SELP   8192u
#define OFF_SELL   12288u
#define OFF_PRED   16384u      // int[51200]
#define OFF_PROBC  221184u     // float[51200]
#define OFF_FH     425984u     // half[1024*256]
#define OFF_AH     950272u     // half[1024*256]
#define OFF_KH     1474560u    // half[2048*256]
#define OFF_S1     2523136u    // float[1024*1024]
#define OFF_S2     6717440u    // float[1024*3072]

// ---------------- init ----------------
__global__ void k_init(int* mi) {
  int t = threadIdx.x;
  if (t < 1024) mi[t] = 0;
}

// ------------- softmax + argmax + prob-of-label -------------
__global__ void k_softmax_argmax(const float* __restrict__ seg, const int* __restrict__ gt,
                                 float* __restrict__ probc, int* __restrict__ pred) {
  int idx = blockIdx.x * 256 + threadIdx.x;
  if (idx >= BATCH * HW) return;
  int b = idx / HW, p = idx - b * HW;
  const float* s = seg + (size_t)b * NCLS * HW + p;
  float mx = s[0]; int am = 0;
  for (int c = 1; c < NCLS; ++c) {
    float v = s[(size_t)c * HW];
    if (v > mx) { mx = v; am = c; }   // first max wins (jnp.argmax)
  }
  float den = 0.f;
  for (int c = 0; c < NCLS; ++c) den += expf(s[(size_t)c * HW] - mx);
  int lab = gt[idx];
  float pc = 0.f;
  if (lab >= 0 && lab < NCLS) pc = expf(s[(size_t)lab * HW] - mx) / den;
  probc[idx] = pc;
  pred[idx]  = am;
}

// ------------- per (batch,class) counts -------------
__global__ void k_count(const int* __restrict__ gt, const int* __restrict__ pred, int* mi) {
  int idx = blockIdx.x * 256 + threadIdx.x;
  if (idx >= BATCH * HW) return;
  int lab = gt[idx];
  if (lab < 0 || lab >= NCLS) return;  // IGNORE / invalid
  int b = idx / HW;
  atomicAdd(&mi[8 + b * NCLS + lab], 1);
  if (pred[idx] != lab) atomicAdd(&mi[64 + b * NCLS + lab], 1);
}

// ------------- mining meta: total, n_view, nhk/nek, base offsets -------------
__global__ void k_meta(int* mi) {
  if (threadIdx.x != 0 || blockIdx.x != 0) return;
  int total = 0;
  for (int bin = 0; bin < BATCH * NCLS; ++bin)
    if (mi[8 + bin] > THRESH) total++;
  mi[1] = total;
  if (total == 0) { mi[0] = 0; return; }
  int n_view = MAXS / total;
  if (n_view > THRESH) n_view = THRESH;
  mi[2] = n_view;
  int base = 0;
  for (int bin = 0; bin < BATCH * NCLS; ++bin) {
    int* bm = &mi[128 + bin * 4];
    int cnt = mi[8 + bin];
    if (cnt > THRESH) {
      int nh = mi[64 + bin], ne = cnt - nh;
      int nhk, nek;
      if (2 * nh >= n_view && 2 * ne >= n_view) { nhk = n_view / 2; nek = n_view - nhk; }
      else if (2 * nh >= n_view)                { nek = ne;         nhk = n_view - nek; }
      else if (2 * ne >= n_view)                { nhk = nh;         nek = n_view - nhk; }
      else                                      { nhk = nh;         nek = ne; }
      if (nhk > nh) nhk = nh;   // numpy [:k] slice clamps
      if (nek > ne) nek = ne;
      bm[0] = 1; bm[1] = nhk; bm[2] = nek; bm[3] = base;
      base += nhk + nek;
    } else {
      bm[0] = 0; bm[1] = 0; bm[2] = 0; bm[3] = base;
    }
  }
  mi[0] = base;  // n_total
}

// ordered-uint encoding: monotone with float order
__device__ __forceinline__ u32 ordkey(float f) {
  u32 u = __float_as_uint(f);
  return (u & 0x80000000u) ? ~u : (u | 0x80000000u);
}

// ------------- iterative stable top-k selection per (b,c) -------------
__global__ void k_select(const int* __restrict__ gt, const int* __restrict__ pred,
                         const float* __restrict__ probc, const int* __restrict__ mi,
                         int* __restrict__ selb, int* __restrict__ selp, int* __restrict__ sell) {
  int bin = blockIdx.x;
  const int* bm = &mi[128 + bin * 4];
  if (!bm[0]) return;
  int nhk = bm[1], nek = bm[2], base = bm[3];
  int b = bin / NCLS, c = bin - b * NCLS;
  int tid = threadIdx.x;
  __shared__ u64 sk[256];
  const int*   gtb = gt    + b * HW;
  const int*   prb = pred  + b * HW;
  const float* pcb = probc + b * HW;

  for (int phase = 0; phase < 2; ++phase) {
    int kcount = (phase == 0) ? nhk : nek;
    u64 last = 0;  // all legit keys > 0
    for (int t = 0; t < kcount; ++t) {
      u64 best = ~0ull;
      for (int p = tid; p < HW; p += 256) {
        if (gtb[p] != c) continue;
        bool ishard = (prb[p] != c);
        if ((phase == 0) != ishard) continue;
        float pv = pcb[p];
        // hard: prob descending (argsort(-prob)); easy: prob ascending; ties -> lower index
        u32 ok = ordkey((phase == 0) ? -pv : pv);
        u64 key = ((u64)ok << 32) | (u32)p;
        if (key > last && key < best) best = key;
      }
      sk[tid] = best; __syncthreads();
      for (int s = 128; s > 0; s >>= 1) {
        if (tid < s && sk[tid + s] < sk[tid]) sk[tid] = sk[tid + s];
        __syncthreads();
      }
      u64 win = sk[0]; __syncthreads();
      if (win == ~0ull) break;
      last = win;
      if (tid == 0) {
        int slot = base + ((phase == 0) ? t : nhk + t);
        selp[slot] = (int)(win & 0xffffffffu);
        selb[slot] = b;
        sell[slot] = c;
      }
    }
    __syncthreads();
  }
}

// ------------- block sum helper -------------
__device__ __forceinline__ float block_sum(float v, float* red) {
  int tid = threadIdx.x;
  red[tid] = v; __syncthreads();
  for (int s = 128; s > 0; s >>= 1) {
    if (tid < s) red[tid] += red[tid + s];
    __syncthreads();
  }
  float r = red[0]; __syncthreads();
  return r;
}

// ------------- gather selected pixels, L2-normalize, convert to f16 -------------
__global__ void k_gather(const float* __restrict__ feats, const float* __restrict__ masks,
                         const int* __restrict__ mi, const int* __restrict__ selb,
                         const int* __restrict__ selp,
                         half_t* __restrict__ F, half_t* __restrict__ A, half_t* __restrict__ K) {
  int s = blockIdx.x, c = threadIdx.x;
  int n = mi[0];
  if (s >= n) {  // zero-pad so WMMA tiles beyond n multiply zeros
    F[(size_t)s * CH + c] = (half_t)0.f;
    A[(size_t)s * CH + c] = (half_t)0.f;
    K[((size_t)s * 2 + 0) * CH + c] = (half_t)0.f;
    K[((size_t)s * 2 + 1) * CH + c] = (half_t)0.f;
    return;
  }
  __shared__ float red[256];
  int b = selb[s], p = selp[s];
  float v = feats[((size_t)b * CH + c) * HW + p];
  float nrm = sqrtf(block_sum(v * v, red));
  F[(size_t)s * CH + c] = (half_t)(v / fmaxf(nrm, 1e-12f));
  float a = masks[(((size_t)(NM - 1) * BATCH + b) * CH + c) * HW + p];
  nrm = sqrtf(block_sum(a * a, red));
  A[(size_t)s * CH + c] = (half_t)(a / fmaxf(nrm, 1e-12f));
  for (int m = 0; m < NKEY; ++m) {
    float km = masks[(((size_t)m * BATCH + b) * CH + c) * HW + p];
    nrm = sqrtf(block_sum(km * km, red));
    K[((size_t)s * 2 + m) * CH + c] = (half_t)(km / fmaxf(nrm, 1e-12f));
  }
}

// ------------- WMMA fragment loaders (CDNA5 ISA 7.12.2) -------------
// A 16x32 f16 (MxK): lane = L, M = L%16; VGPR v holds K = (v>=4?16:0) + (L/16)*8 + (v%4)*2, +1
__device__ __forceinline__ v16h frag_a(const half_t* __restrict__ F, int row0, int kk, int lane) {
  int m  = lane & 15;
  int kl = (lane >> 4) << 3;     // 0 or 8
  const float* rp = (const float*)(F + (size_t)(row0 + m) * CH + kk);
  v16h a;
  float* af = (float*)&a;
  #pragma unroll
  for (int v = 0; v < 8; ++v) {
    int Kh = ((v & 4) << 2) + kl + ((v & 3) << 1);  // even half index in [0,32)
    af[v] = rp[Kh >> 1];
  }
  return a;
}
// B 32x16 f16 (KxN): N = L%16; VGPR v holds K = (v>=4?16:0) + (L/16)*8 + (v%4)*2, +1
__device__ __forceinline__ v16h frag_b(const half_t* __restrict__ F, int col0, int kk, int lane) {
  int nn = lane & 15;
  int kl = (lane >> 4) << 3;     // 0 or 8
  const float* rp = (const float*)(F + (size_t)(col0 + nn) * CH + kk);
  v16h b;
  float* bf = (float*)&b;
  #pragma unroll
  for (int v = 0; v < 8; ++v) bf[v] = rp[(((v & 4) << 2) + kl + ((v & 3) << 1)) >> 1];
  return b;
}

// ------------- S = A * B^T via v_wmma_f32_16x16x32_f16, one wave per 16x16 tile -------------
__global__ void k_gemm_nt(const half_t* __restrict__ A, const half_t* __restrict__ B,
                          float* __restrict__ C, int ldc, int coloff) {
  int lane = threadIdx.x;
  int c0 = blockIdx.x * 16, r0 = blockIdx.y * 16;
  v8f acc = {};
  #pragma unroll
  for (int kk = 0; kk < CH; kk += 32) {
    v16h a = frag_a(A, r0, kk, lane);
    v16h b = frag_b(B, c0, kk, lane);
    acc = __builtin_amdgcn_wmma_f32_16x16x32_f16(false, a, false, b, (short)0, acc,
                                                 false, false);
  }
  // C/D layout: N = lane%16; VGPR r -> M = r + (lane<16 ? 0 : 8)
  int nn = lane & 15;
  int mb = (lane < 16) ? 0 : 8;
  #pragma unroll
  for (int r = 0; r < 8; ++r)
    C[(size_t)(r0 + mb + r) * ldc + coloff + c0 + nn] = acc[r];
}

// ------------- feature contrastive loss, one block per anchor row -------------
__global__ void k_feat_loss(const float* __restrict__ S, const int* __restrict__ sell,
                            const int* __restrict__ mi, float* acc) {
  int i = blockIdx.x;
  int n = mi[0];
  if (i >= n) return;
  int tid = threadIdx.x;
  __shared__ float red[256];
  const float* row = S + (size_t)i * NPAD;
  int li = sell[i];
  float lmax = -1e30f;
  for (int j = tid; j < n; j += 256) lmax = fmaxf(lmax, row[j] * INV_TEMP);
  red[tid] = lmax; __syncthreads();
  for (int s = 128; s > 0; s >>= 1) {
    if (tid < s) red[tid] = fmaxf(red[tid], red[tid + s]);
    __syncthreads();
  }
  float M = red[0]; __syncthreads();
  float se = 0.f, sp = 0.f, np_ = 0.f;
  for (int j = tid; j < n; j += 256) {
    float l = row[j] * INV_TEMP - M;
    if (j != i) {
      se += expf(l);
      if (sell[j] == li) { sp += l; np_ += 1.f; }
    }
  }
  se  = block_sum(se,  red);
  sp  = block_sum(sp,  red);
  np_ = block_sum(np_, red);
  if (tid == 0) {
    float logZ = logf(se + EPS_);
    float mlpp = (sp - np_ * logZ) / (np_ + EPS_);
    atomicAdd(acc, -T_OVER_BT * mlpp);
  }
}

// ------------- mask contrastive loss (anchors vs [anchors | keys]) -------------
__global__ void k_mask_loss(const float* __restrict__ S2, const int* __restrict__ sell,
                            const int* __restrict__ mi, float* acc) {
  int i = blockIdx.x;
  int n = mi[0];
  if (i >= n) return;
  int tid = threadIdx.x;
  __shared__ float red[256];
  const float* row = S2 + (size_t)i * LDC2;
  int li = sell[i];
  float lmax = -1e30f;
  for (int j = tid; j < n; j += 256)     lmax = fmaxf(lmax, row[j] * INV_TEMP);
  for (int q = tid; q < 2 * n; q += 256) lmax = fmaxf(lmax, row[NPAD + q] * INV_TEMP);
  red[tid] = lmax; __syncthreads();
  for (int s = 128; s > 0; s >>= 1) {
    if (tid < s) red[tid] = fmaxf(red[tid], red[tid + s]);
    __syncthreads();
  }
  float M = red[0]; __syncthreads();
  float se = 0.f, sp = 0.f, nt = 0.f;
  // anchor-vs-anchor block: tot = same & off-diag; neg = ~tot (diagonal is negative)
  for (int j = tid; j < n; j += 256) {
    float l = row[j] * INV_TEMP - M;
    bool tot = (sell[j] == li) && (j != i);
    if (tot) { sp += l; nt += 1.f; } else se += expf(l);
  }
  // key block: key_label = tile(same,(1,k)) -> column q uses label[q % n]
  for (int q = tid; q < 2 * n; q += 256) {
    float l = row[NPAD + q] * INV_TEMP - M;
    bool tot = (sell[q % n] == li);
    if (tot) { sp += l; nt += 1.f; } else se += expf(l);
  }
  se = block_sum(se, red);
  sp = block_sum(sp, red);
  nt = block_sum(nt, red);
  if (tid == 0) {
    float logZ = logf(se + EPS_);
    float mlp = (sp - nt * logZ) / (nt + EPS_);
    atomicAdd(acc + 1, -T_OVER_BT * mlp);
  }
}

// ------------- finalize -------------
__global__ void k_finalize(const float* __restrict__ acc, const int* __restrict__ mi,
                           float* __restrict__ out) {
  if (threadIdx.x != 0 || blockIdx.x != 0) return;
  int n = mi[0];
  if (n > 0) {
    out[0] = FEAT_W_ * acc[0] / (float)n;
    out[1] = MASK_W_ * acc[1] / (float)n;
  } else {
    out[0] = 0.f; out[1] = 0.f;
  }
}

extern "C" void kernel_launch(void* const* d_in, const int* in_sizes, int n_in,
                              void* d_out, int out_size, void* d_ws, size_t ws_size,
                              hipStream_t stream) {
  (void)in_sizes; (void)n_in; (void)out_size; (void)ws_size;
  const float* feats = (const float*)d_in[0];   // (B,256,H,W) f32
  const float* seg   = (const float*)d_in[1];   // (B,19,H,W)  f32
  const int*   gt    = (const int*)d_in[2];     // (B,1,H,W)   i32
  const float* masks = (const float*)d_in[3];   // (3,B,256,H,W) f32

  char* ws = (char*)d_ws;
  int*    mi    = (int*)(ws + OFF_META);
  int*    selb  = (int*)(ws + OFF_SELB);
  int*    selp  = (int*)(ws + OFF_SELP);
  int*    sell  = (int*)(ws + OFF_SELL);
  int*    pred  = (int*)(ws + OFF_PRED);
  float*  probc = (float*)(ws + OFF_PROBC);
  half_t* Fh    = (half_t*)(ws + OFF_FH);
  half_t* Ah    = (half_t*)(ws + OFF_AH);
  half_t* Kh    = (half_t*)(ws + OFF_KH);
  float*  S1    = (float*)(ws + OFF_S1);
  float*  S2    = (float*)(ws + OFF_S2);
  float*  acc   = (float*)mi + 320;

  int tot = BATCH * HW;
  k_init<<<1, 1024, 0, stream>>>(mi);
  k_softmax_argmax<<<(tot + 255) / 256, 256, 0, stream>>>(seg, gt, probc, pred);
  k_count<<<(tot + 255) / 256, 256, 0, stream>>>(gt, pred, mi);
  k_meta<<<1, 1, 0, stream>>>(mi);
  k_select<<<BATCH * NCLS, 256, 0, stream>>>(gt, pred, probc, mi, selb, selp, sell);
  k_gather<<<NPAD, 256, 0, stream>>>(feats, masks, mi, selb, selp, Fh, Ah, Kh);
  dim3 g1(NPAD / 16, NPAD / 16);
  k_gemm_nt<<<g1, 32, 0, stream>>>(Fh, Fh, S1, NPAD, 0);          // F F^T
  k_gemm_nt<<<g1, 32, 0, stream>>>(Ah, Ah, S2, LDC2, 0);          // anchors vs anchors
  dim3 g2(KROWS / 16, NPAD / 16);
  k_gemm_nt<<<g2, 32, 0, stream>>>(Ah, Kh, S2, LDC2, NPAD);       // anchors vs keys
  k_feat_loss<<<NPAD, 256, 0, stream>>>(S1, sell, mi, acc);
  k_mask_loss<<<NPAD, 256, 0, stream>>>(S2, sell, mi, acc);
  k_finalize<<<1, 1, 0, stream>>>(acc, mi, (float*)d_out);
}
